// DepthWiseSeparableAttention_53197464928518
// MI455X (gfx1250) — matrix operand order repaired and checked by reading, not compiled, not run on hardware
//
#include <hip/hip_runtime.h>
#include <hip/hip_bf16.h>

// Shapes: B=8, N=1024, C=512, HEADS=8, DH=64.  SCALE = 1/8 folded into Q.
// Workspace layout (bf16 = unsigned short), E = 8192*512 elements each:
//   [0,E)   xn   : LayerNorm output, (B*N, C) row-major
//   [E,2E)  q    : (B,H,N,DH), pre-scaled by 1/8
//   [2E,3E) k    : (B,H,N,DH)
//   [3E,4E) vT   : (B,H,DH,N)  (transposed so PV B-fragments are contiguous)
//   [4E,5E) ao   : attention output, (B*N, C) row-major
// Total 40 MB of d_ws.

typedef __bf16 bhalf;
typedef __attribute__((ext_vector_type(8)))  bhalf  bhalf8;
typedef __attribute__((ext_vector_type(16))) bhalf  bhalf16;
typedef __attribute__((ext_vector_type(8)))  float  v8f;
typedef __attribute__((ext_vector_type(8)))  unsigned short us8;

static __device__ __forceinline__ bhalf16 bcat(bhalf8 lo, bhalf8 hi) {
  return __builtin_shufflevector(lo, hi, 0,1,2,3,4,5,6,7,8,9,10,11,12,13,14,15);
}
static __device__ __forceinline__ unsigned short f2bf(float f) {
  union { float f; unsigned u; } c; c.f = f;
  return (unsigned short)((c.u + 0x7FFFu + ((c.u >> 16) & 1u)) >> 16);
}
static __device__ __forceinline__ v8f wmma_bf16(bhalf16 a, bhalf16 b, v8f c) {
  // D = A(16x32 bf16) x B(32x16 bf16) + C(16x16 f32)
  return __builtin_amdgcn_wmma_f32_16x16x32_bf16(false, a, false, b, (short)0, c,
                                                 false, false);
}

// CDNA5 async copy global->LDS (ASYNCcnt path).  VDST = LDS byte address
// (low 32 bits of the flat shared pointer), VADDR = 64-bit global address.
static __device__ __forceinline__ void async_ld_b128(unsigned lds_off,
                                                     const void* gptr) {
  asm volatile("global_load_async_to_lds_b128 %0, %1, off"
               :: "v"(lds_off), "v"(gptr) : "memory");
}
static __device__ __forceinline__ void wait_async0() {
  asm volatile("s_wait_asynccnt 0" ::: "memory");
}
static __device__ __forceinline__ unsigned lds_addr(const void* shared_ptr) {
  return (unsigned)(uintptr_t)shared_ptr;  // flat LDS addr低32 = LDS offset
}

// ---------------------------------------------------------------------------
// Kernel 1: LayerNorm (eps=1e-6) -> bf16.  One wave32 per 512-wide row.
// ---------------------------------------------------------------------------
__global__ __launch_bounds__(256) void k_ln(const float* __restrict__ x,
                                            const float* __restrict__ g,
                                            const float* __restrict__ be,
                                            unsigned short* __restrict__ xn) {
  const int row  = blockIdx.x * 8 + (threadIdx.x >> 5);
  const int lane = threadIdx.x & 31;
  const float4* xr = (const float4*)(x + (size_t)row * 512);
  float4 v[4];
  float s = 0.f, ss = 0.f;
#pragma unroll
  for (int i = 0; i < 4; ++i) {
    v[i] = xr[lane + 32 * i];
    s  += v[i].x + v[i].y + v[i].z + v[i].w;
    ss += v[i].x * v[i].x + v[i].y * v[i].y + v[i].z * v[i].z + v[i].w * v[i].w;
  }
#pragma unroll
  for (int m = 1; m < 32; m <<= 1) {
    s  += __shfl_xor(s, m, 32);
    ss += __shfl_xor(ss, m, 32);
  }
  const float mu  = s * (1.f / 512.f);
  const float inv = rsqrtf(ss * (1.f / 512.f) - mu * mu + 1e-6f);
  unsigned short* orow = xn + (size_t)row * 512;
#pragma unroll
  for (int i = 0; i < 4; ++i) {
    float4 gv = ((const float4*)g)[lane + 32 * i];
    float4 bv = ((const float4*)be)[lane + 32 * i];
    ushort4 o;
    o.x = f2bf((v[i].x - mu) * inv * gv.x + bv.x);
    o.y = f2bf((v[i].y - mu) * inv * gv.y + bv.y);
    o.z = f2bf((v[i].z - mu) * inv * gv.z + bv.z);
    o.w = f2bf((v[i].w - mu) * inv * gv.w + bv.w);
    ((ushort4*)orow)[lane + 32 * i] = o;
  }
}

// ---------------------------------------------------------------------------
// Kernel 2: QKV GEMM (8192x512 @ 512x1536^T) bf16 WMMA, BN affine folded,
// epilogue scatters q (pre-scaled by 1/8), k, vT.
// Workgroup tile 128(M) x 64(N), 8 waves = 4x2, each wave 32x32 = 2x2 WMMA.
// A tile staged via CDNA5 async global->LDS copies.
// ---------------------------------------------------------------------------
__global__ __launch_bounds__(256) void k_qkv(
    const unsigned short* __restrict__ xn, const float* __restrict__ W,
    const float* __restrict__ qbias, const float* __restrict__ bn_g,
    const float* __restrict__ bn_b, const float* __restrict__ bn_m,
    const float* __restrict__ bn_v, unsigned short* __restrict__ qout,
    unsigned short* __restrict__ kout, unsigned short* __restrict__ vtout) {
  __shared__ __align__(16) unsigned short As[128 * 32];
  __shared__ __align__(16) unsigned short Bs[64 * 32];
  const int m0 = blockIdx.x * 128;
  const int o0 = blockIdx.y * 64;
  const int tid = threadIdx.x, lane = tid & 31, wave = tid >> 5;
  const int wm = wave >> 1, wn = wave & 1;
  const int lm = lane & 15, hb = lane >> 4;
  const int kb = hb * 8;
  v8f c[2][2] = {};

  for (int k0 = 0; k0 < 512; k0 += 32) {
    {  // stage A tile: async global->LDS, 2 threads per row, 2x16B each
      const int r = tid >> 1, hf = tid & 1;
      const unsigned short* gsrc = xn + (size_t)(m0 + r) * 512 + k0 + hf * 16;
      const unsigned l0 = lds_addr(As + r * 32 + hf * 16);
      async_ld_b128(l0, gsrc);
      async_ld_b128(l0 + 16, gsrc + 8);
      if (k0 + 32 < 512)
        __builtin_prefetch(gsrc + 16, 0, 3);
    }
    {  // stage B tile: W fp32 -> bf16, kept row-major [o][k] (B-fragment friendly)
      const int ol = tid >> 2, kk = (tid & 3) * 8;
      const float4* wsrc = (const float4*)(W + (size_t)(o0 + ol) * 512 + k0 + kk);
      float4 w0 = wsrc[0], w1 = wsrc[1];
      us8 p;
      p[0] = f2bf(w0.x); p[1] = f2bf(w0.y); p[2] = f2bf(w0.z); p[3] = f2bf(w0.w);
      p[4] = f2bf(w1.x); p[5] = f2bf(w1.y); p[6] = f2bf(w1.z); p[7] = f2bf(w1.w);
      *(us8*)(Bs + ol * 32 + kk) = p;
    }
    wait_async0();
    __syncthreads();
    bhalf16 af[2], bf2[2];
#pragma unroll
    for (int i = 0; i < 2; ++i) {
      const int ar = wm * 32 + i * 16 + lm;   // A: row per lane, 16 consecutive K
      af[i] = bcat(*(const bhalf8*)(As + ar * 32 + hb * 16),
                   *(const bhalf8*)(As + ar * 32 + hb * 16 + 8));
      const int br = wn * 32 + i * 16 + lm;   // B: col per lane, 16 consecutive K
      bf2[i] = bcat(*(const bhalf8*)(Bs + br * 32 + hb * 16),
                    *(const bhalf8*)(Bs + br * 32 + hb * 16 + 8));
    }
    c[0][0] = wmma_bf16(af[0], bf2[0], c[0][0]);
    c[0][1] = wmma_bf16(af[0], bf2[1], c[0][1]);
    c[1][0] = wmma_bf16(af[1], bf2[0], c[1][0]);
    c[1][1] = wmma_bf16(af[1], bf2[1], c[1][1]);
    __syncthreads();
  }

  // Epilogue: BN affine fold, scatter to q (x SCALE), k, vT
#pragma unroll
  for (int ni = 0; ni < 2; ++ni) {
    const int o = o0 + wn * 32 + ni * 16 + lm;
    const float sc = bn_g[o] * rsqrtf(bn_v[o] + 1e-5f);
    const float sh = (qbias[o] - bn_m[o]) * sc + bn_b[o];
    const int sect = o >> 9, oo = o & 511, hh = oo >> 6, dd = oo & 63;
#pragma unroll
    for (int mi = 0; mi < 2; ++mi) {
#pragma unroll
      for (int r = 0; r < 8; ++r) {
        const int row = m0 + wm * 32 + mi * 16 + r + 8 * hb;
        const int bb = row >> 10, n = row & 1023;
        const float val = c[mi][ni][r] * sc + sh;
        if (sect == 0) {
          qout[((size_t)(bb * 8 + hh) * 1024 + n) * 64 + dd] = f2bf(val * 0.125f);
        } else if (sect == 1) {
          kout[((size_t)(bb * 8 + hh) * 1024 + n) * 64 + dd] = f2bf(val);
        } else {
          vtout[((size_t)(bb * 8 + hh) * 64 + dd) * 1024 + n] = f2bf(val);
        }
      }
    }
  }
}

// ---------------------------------------------------------------------------
// Kernel 3: flash attention per (b,h).  Grid = 64 heads x 8 query blocks.
// Each wave owns 16 query rows, sweeps keys in 64-wide blocks with online
// softmax.  P is re-laid out C->A through a wave-private LDS slab.
// (The pooled/conv "local bias" is a per-row constant under softmax => no-op.)
// ---------------------------------------------------------------------------
__global__ __launch_bounds__(256) void k_attn(
    const unsigned short* __restrict__ q, const unsigned short* __restrict__ k,
    const unsigned short* __restrict__ vT, unsigned short* __restrict__ ao) {
  __shared__ __align__(16) unsigned short P[8 * 16 * 64];  // 16 KB
  const int bh = blockIdx.x >> 3, qb = blockIdx.x & 7;
  const int wave = threadIdx.x >> 5, lane = threadIdx.x & 31;
  const int lm = lane & 15, hb = lane >> 4, kb = hb * 8;
  const int qrow0 = qb * 128 + wave * 16;
  const size_t base = (size_t)bh * 1024 * 64;
  unsigned short* Pw = P + wave * 1024;

  // Q A-fragments (d-split K=32 steps), loaded once
  const unsigned short* qrow = q + base + (size_t)(qrow0 + lm) * 64;
  bhalf16 aq[2];
#pragma unroll
  for (int ks = 0; ks < 2; ++ks)
    aq[ks] = bcat(*(const bhalf8*)(qrow + ks * 32 + hb * 16),
                  *(const bhalf8*)(qrow + ks * 32 + hb * 16 + 8));

  v8f oacc[4] = {};
  float rmax[8], rsum[8];
#pragma unroll
  for (int r = 0; r < 8; ++r) { rmax[r] = -1e30f; rsum[r] = 0.f; }

  for (int kb0 = 0; kb0 < 1024; kb0 += 64) {
    // S = Q K^T : 4 key tiles x 2 d-steps
    v8f s[4] = {};
#pragma unroll
    for (int ks = 0; ks < 2; ++ks) {
#pragma unroll
      for (int t = 0; t < 4; ++t) {
        const unsigned short* kp =
            k + base + (size_t)(kb0 + t * 16 + lm) * 64 + ks * 32 + hb * 16;
        s[t] = wmma_bf16(aq[ks], bcat(*(const bhalf8*)kp, *(const bhalf8*)(kp + 8)),
                         s[t]);
      }
    }
    // Online softmax update (per-row over 64 keys; rows live in half-waves)
#pragma unroll
    for (int r = 0; r < 8; ++r) {
      float m4 = fmaxf(fmaxf(s[0][r], s[1][r]), fmaxf(s[2][r], s[3][r]));
#pragma unroll
      for (int m = 1; m < 16; m <<= 1) m4 = fmaxf(m4, __shfl_xor(m4, m, 32));
      const float nm = fmaxf(rmax[r], m4);
      const float corr = __expf(rmax[r] - nm);
      rmax[r] = nm;
      float psum = 0.f;
#pragma unroll
      for (int t = 0; t < 4; ++t) {
        const float p = __expf(s[t][r] - nm);
        s[t][r] = p;
        psum += p;
      }
#pragma unroll
      for (int m = 1; m < 16; m <<= 1) psum += __shfl_xor(psum, m, 32);
      rsum[r] = rsum[r] * corr + psum;
#pragma unroll
      for (int t = 0; t < 4; ++t) oacc[t][r] *= corr;
      const int mrow = r + 8 * hb;
#pragma unroll
      for (int t = 0; t < 4; ++t) Pw[mrow * 64 + t * 16 + lm] = f2bf(s[t][r]);
    }
    // O += P V : A from LDS (row-major P), B from vT (contiguous keys)
#pragma unroll
    for (int ks = 0; ks < 2; ++ks) {
      const bhalf16 ap = bcat(*(const bhalf8*)(Pw + lm * 64 + ks * 32 + hb * 16),
                              *(const bhalf8*)(Pw + lm * 64 + ks * 32 + hb * 16 + 8));
#pragma unroll
      for (int t = 0; t < 4; ++t) {
        const unsigned short* vp =
            vT + base + (size_t)(t * 16 + lm) * 1024 + kb0 + ks * 32 + hb * 16;
        oacc[t] = wmma_bf16(ap, bcat(*(const bhalf8*)vp, *(const bhalf8*)(vp + 8)),
                            oacc[t]);
      }
    }
  }
  // Epilogue: normalize, write (B,N,C) bf16 for the projection GEMM
  const int bb = bh >> 3, hh = bh & 7;
#pragma unroll
  for (int r = 0; r < 8; ++r) {
    const int n = qrow0 + r + 8 * hb;
    const float inv = 1.0f / rsum[r];
    unsigned short* dst = ao + (size_t)(bb * 1024 + n) * 512 + hh * 64;
#pragma unroll
    for (int t = 0; t < 4; ++t) dst[t * 16 + lm] = f2bf(oacc[t][r] * inv);
  }
}

// ---------------------------------------------------------------------------
// Kernel 4: projection GEMM (8192x512 @ 512x512^T) + bias + residual, fp32 out
// ---------------------------------------------------------------------------
__global__ __launch_bounds__(256) void k_proj(
    const unsigned short* __restrict__ ao, const float* __restrict__ W,
    const float* __restrict__ pb, const float* __restrict__ x,
    float* __restrict__ out) {
  __shared__ __align__(16) unsigned short As[128 * 32];
  __shared__ __align__(16) unsigned short Bs[64 * 32];
  const int m0 = blockIdx.x * 128;
  const int o0 = blockIdx.y * 64;
  const int tid = threadIdx.x, lane = tid & 31, wave = tid >> 5;
  const int wm = wave >> 1, wn = wave & 1;
  const int lm = lane & 15, hb = lane >> 4, kb = hb * 8;
  v8f c[2][2] = {};

  for (int k0 = 0; k0 < 512; k0 += 32) {
    {  // stage A tile via async copies
      const int r = tid >> 1, hf = tid & 1;
      const unsigned short* gsrc = ao + (size_t)(m0 + r) * 512 + k0 + hf * 16;
      const unsigned l0 = lds_addr(As + r * 32 + hf * 16);
      async_ld_b128(l0, gsrc);
      async_ld_b128(l0 + 16, gsrc + 8);
      if (k0 + 32 < 512)
        __builtin_prefetch(gsrc + 16, 0, 3);
    }
    {
      const int ol = tid >> 2, kk = (tid & 3) * 8;
      const float4* wsrc = (const float4*)(W + (size_t)(o0 + ol) * 512 + k0 + kk);
      float4 w0 = wsrc[0], w1 = wsrc[1];
      us8 p;
      p[0] = f2bf(w0.x); p[1] = f2bf(w0.y); p[2] = f2bf(w0.z); p[3] = f2bf(w0.w);
      p[4] = f2bf(w1.x); p[5] = f2bf(w1.y); p[6] = f2bf(w1.z); p[7] = f2bf(w1.w);
      *(us8*)(Bs + ol * 32 + kk) = p;
    }
    wait_async0();
    __syncthreads();
    bhalf16 af[2], bf2[2];
#pragma unroll
    for (int i = 0; i < 2; ++i) {
      const int ar = wm * 32 + i * 16 + lm;
      af[i] = bcat(*(const bhalf8*)(As + ar * 32 + hb * 16),
                   *(const bhalf8*)(As + ar * 32 + hb * 16 + 8));
      const int br = wn * 32 + i * 16 + lm;
      bf2[i] = bcat(*(const bhalf8*)(Bs + br * 32 + hb * 16),
                    *(const bhalf8*)(Bs + br * 32 + hb * 16 + 8));
    }
    c[0][0] = wmma_bf16(af[0], bf2[0], c[0][0]);
    c[0][1] = wmma_bf16(af[0], bf2[1], c[0][1]);
    c[1][0] = wmma_bf16(af[1], bf2[0], c[1][0]);
    c[1][1] = wmma_bf16(af[1], bf2[1], c[1][1]);
    __syncthreads();
  }

#pragma unroll
  for (int ni = 0; ni < 2; ++ni) {
    const int o = o0 + wn * 32 + ni * 16 + lm;
    const float bias = pb[o];
#pragma unroll
    for (int mi = 0; mi < 2; ++mi) {
#pragma unroll
      for (int r = 0; r < 8; ++r) {
        const int row = m0 + wm * 32 + mi * 16 + r + 8 * hb;
        out[(size_t)row * 512 + o] = c[mi][ni][r] + bias + x[(size_t)row * 512 + o];
      }
    }
  }
}

// ---------------------------------------------------------------------------
extern "C" void kernel_launch(void* const* d_in, const int* in_sizes, int n_in,
                              void* d_out, int out_size, void* d_ws,
                              size_t ws_size, hipStream_t stream) {
  (void)in_sizes; (void)n_in; (void)out_size; (void)ws_size;
  const float* x      = (const float*)d_in[0];
  const float* ln_g   = (const float*)d_in[1];
  const float* ln_b   = (const float*)d_in[2];
  const float* qkv_w  = (const float*)d_in[3];
  const float* qkv_b  = (const float*)d_in[4];
  const float* bn_g   = (const float*)d_in[5];
  const float* bn_b   = (const float*)d_in[6];
  const float* bn_m   = (const float*)d_in[7];
  const float* bn_v   = (const float*)d_in[8];
  // d_in[9] (conv_w): per-row constant bias under softmax => exact no-op.
  const float* proj_w = (const float*)d_in[10];
  const float* proj_b = (const float*)d_in[11];

  unsigned short* ws = (unsigned short*)d_ws;
  const size_t E = (size_t)8192 * 512;
  unsigned short* xn = ws;
  unsigned short* qq = ws + E;
  unsigned short* kk = ws + 2 * E;
  unsigned short* vT = ws + 3 * E;
  unsigned short* ao = ws + 4 * E;

  k_ln<<<1024, 256, 0, stream>>>(x, ln_g, ln_b, xn);
  k_qkv<<<dim3(64, 24), 256, 0, stream>>>(xn, qkv_w, qkv_b, bn_g, bn_b, bn_m,
                                          bn_v, qq, kk, vT);
  k_attn<<<512, 256, 0, stream>>>(qq, kk, vT, ao);
  k_proj<<<dim3(64, 8), 256, 0, stream>>>(ao, proj_w, proj_b, x, (float*)d_out);
}
